// SWinTransformerBlock_25580825215498
// MI455X (gfx1250) — matrix operand order repaired and checked by reading, not compiled, not run on hardware
//
#include <hip/hip_runtime.h>

// ---------------------------------------------------------------------------
// Swin block on gfx1250: all matmuls on V_WMMA_F32_16X16X32_BF16 (f32 accum),
// GEMM tiles streamed with GLOBAL_LOAD_ASYNC_TO_LDS_B128 + double-buffered LDS,
// cross-lane reductions via single-instruction ds_swizzle XOR butterflies.
// B=32 H=W=56 DIM=512 NH=16 HD=32 WS=7 SS=3 N=49, rows M = B*L = 100352.
// Workspace (~537 MB): bf16 weightsT + 2 activation buffers + bias/mask table.
// ---------------------------------------------------------------------------

typedef __attribute__((ext_vector_type(16))) __bf16 bf16x16;
typedef __attribute__((ext_vector_type(8)))  __bf16 bf16x8;
typedef __attribute__((ext_vector_type(8)))  float  f32x8;

#define BCAT(lo, hi) __builtin_shufflevector(lo, hi, 0,1,2,3,4,5,6,7,8,9,10,11,12,13,14,15)

__device__ __forceinline__ f32x8 wmma_bf16f32(bf16x16 a, bf16x16 b, f32x8 c) {
  // emits v_wmma_f32_16x16x32_bf16
  return __builtin_amdgcn_wmma_f32_16x16x32_bf16(false, a, false, b, (short)0, c, false, false);
}

__device__ __forceinline__ bf16x8 bzero8() {
  bf16x8 z;
#pragma unroll
  for (int i = 0; i < 8; ++i) z[i] = (__bf16)0.0f;
  return z;
}
__device__ __forceinline__ f32x8 fzero8() {
  f32x8 z;
#pragma unroll
  for (int i = 0; i < 8; ++i) z[i] = 0.0f;
  return z;
}

// Single ds_swizzle_b32 XOR butterfly (group-of-32 mode: and=0x1f, or=0, xor=M).
// M < 16 stays within a 16-lane half; M up to 16 spans the full wave32.
template <int M>
__device__ __forceinline__ float xorSwz(float v) {
  int i = __builtin_amdgcn_ds_swizzle(__builtin_bit_cast(int, v), (M << 10) | 0x1f);
  return __builtin_bit_cast(float, i);
}

// Async global -> LDS copy, 16B per lane (GVS mode: 64-bit SGPR base + 32-bit
// per-lane byte offset). Tracked by ASYNCcnt.
__device__ __forceinline__ void async_ld_b128(unsigned lds_addr, unsigned gofs,
                                              const void* base) {
  asm volatile("global_load_async_to_lds_b128 %0, %1, %2"
               :: "v"(lds_addr), "v"(gofs), "s"(base) : "memory");
}
__device__ __forceinline__ void wait_async0() {
  asm volatile("s_wait_asynccnt 0x0" ::: "memory");
}
__device__ __forceinline__ unsigned lds_off(const void* p) {
  return (unsigned)(uintptr_t)p;   // low 32 bits of shared aperture = LDS offset
}

// region id along one axis of the shifted 56-pixel image (for the shift mask)
__device__ __forceinline__ int regionOf(int g) { return (g < 49) ? 0 : ((g < 53) ? 1 : 2); }

// ---------------------------------------------------------------------------
// Weight cast + transpose: W (K x N, f32 row-major) -> Wt (N x K, bf16)
// ---------------------------------------------------------------------------
__global__ void castT_kernel(const float* __restrict__ W, __bf16* __restrict__ Wt,
                             int K, int N) {
  int n = blockIdx.x * 16 + threadIdx.x;
  int k = blockIdx.y * 16 + threadIdx.y;
  if (k < K && n < N) Wt[(size_t)n * K + k] = (__bf16)W[(size_t)k * N + n];
}

// ---------------------------------------------------------------------------
// Combined rel-pos-bias + shift-mask table:
// comb[(wIdx*16+head)][i(64)][j(64)] ; pad entries = -1e30 (softmax zeros).
// 64*16*4096 f32 = 16.8 MB -> L2 resident.
// ---------------------------------------------------------------------------
__global__ __launch_bounds__(256) void biasmask_kernel(const float* __restrict__ relTab,
                                                       float* __restrict__ comb) {
  int wIdx = blockIdx.x >> 4;
  int head = blockIdx.x & 15;
  int wh = wIdx >> 3, ww = wIdx & 7;
  for (int idx = threadIdx.x; idx < 4096; idx += 256) {
    int i = idx >> 6, j = idx & 63;
    float v = -1e30f;
    if (i < 49 && j < 49) {
      int ih = i / 7, iw = i % 7, jh = j / 7, jw = j % 7;
      int ridx = (ih - jh + 6) * 13 + (iw - jw + 6);
      float bias = relTab[ridx * 16 + head];
      int ri = regionOf(wh * 7 + ih) * 3 + regionOf(ww * 7 + iw);
      int rj = regionOf(wh * 7 + jh) * 3 + regionOf(ww * 7 + jw);
      v = bias + ((ri == rj) ? 0.0f : -100.0f);
    }
    comb[(size_t)blockIdx.x * 4096 + idx] = v;
  }
}

// ---------------------------------------------------------------------------
// LayerNorm. MODE 0: fused cyclic-shift + window-partition gather, output row
// order = (window, token). MODE 1: identity row order. One wave per row.
// ---------------------------------------------------------------------------
template <int MODE>
__global__ __launch_bounds__(256) void ln_kernel(const float* __restrict__ x,
                                                 const float* __restrict__ g,
                                                 const float* __restrict__ bta,
                                                 __bf16* __restrict__ out) {
  int row = blockIdx.x * 8 + (threadIdx.x >> 5);
  int lane = threadIdx.x & 31;
  size_t srcRow;
  if (MODE == 0) {
    int w = row / 49, t = row % 49;
    int bi = w >> 6, wi = w & 63;
    int wh = wi >> 3, ww = wi & 7;
    int sr = wh * 7 + t / 7, sc = ww * 7 + t % 7;   // coords in shifted image
    int gr = sr + 3; if (gr >= 56) gr -= 56;        // source coords in x (roll -3)
    int gc = sc + 3; if (gc >= 56) gc -= 56;
    srcRow = (size_t)bi * 3136 + gr * 56 + gc;
  } else {
    srcRow = row;
  }
  const float4* src = (const float4*)(x + srcRow * 512);
  float4 v[4];
  float sum = 0.f, sq = 0.f;
#pragma unroll
  for (int i = 0; i < 4; ++i) {
    v[i] = src[lane + i * 32];
    sum += v[i].x + v[i].y + v[i].z + v[i].w;
    sq  += v[i].x * v[i].x + v[i].y * v[i].y + v[i].z * v[i].z + v[i].w * v[i].w;
  }
  sum += xorSwz<16>(sum); sum += xorSwz<8>(sum); sum += xorSwz<4>(sum);
  sum += xorSwz<2>(sum);  sum += xorSwz<1>(sum);
  sq  += xorSwz<16>(sq);  sq  += xorSwz<8>(sq);  sq  += xorSwz<4>(sq);
  sq  += xorSwz<2>(sq);   sq  += xorSwz<1>(sq);
  float mu = sum * (1.0f / 512.0f);
  float rstd = rsqrtf(sq * (1.0f / 512.0f) - mu * mu + 1e-5f);
  const float4* g4 = (const float4*)g;
  const float4* b4 = (const float4*)bta;
  __bf16* orow = out + (size_t)row * 512;
#pragma unroll
  for (int i = 0; i < 4; ++i) {
    float4 gv = g4[lane + i * 32], bv = b4[lane + i * 32];
    int e = (lane + i * 32) * 4;
    orow[e + 0] = (__bf16)((v[i].x - mu) * rstd * gv.x + bv.x);
    orow[e + 1] = (__bf16)((v[i].y - mu) * rstd * gv.y + bv.y);
    orow[e + 2] = (__bf16)((v[i].z - mu) * rstd * gv.z + bv.z);
    orow[e + 3] = (__bf16)((v[i].w - mu) * rstd * gv.w + bv.w);
  }
}

// ---------------------------------------------------------------------------
// WMMA GEMM: C(M x N) = A(M x K, bf16) * Wt(N x K, bf16)^T, f32 accumulate.
// Block: 256 thr / 8 waves, 128x128 tile (wave grid 4M x 2N, 32x64 per wave).
// K staged 32 at a time via async LDS loads, double-buffered (1 barrier/step).
// EPI: 0 = +bias, q-cols scaled -> bf16 (qkv)
//      1 = +bias + window-reverse/roll scatter + residual -> f32 d_out (proj)
//      2 = gelu(+bias) -> bf16 (fc1)
//      3 = +bias accumulated into f32 d_out (fc2)
// ---------------------------------------------------------------------------
template <int EPI>
__global__ __launch_bounds__(256) void gemm_bf16(const __bf16* __restrict__ A,
                                                 const __bf16* __restrict__ Wt,
                                                 const float* __restrict__ bias,
                                                 int M, int N, int K,
                                                 __bf16* __restrict__ outB,
                                                 float* __restrict__ outF,
                                                 const float* __restrict__ resid) {
  __shared__ alignas(16) __bf16 As[2][128 * 40];   // 128 rows x 32 K, +8 pad
  __shared__ alignas(16) __bf16 Bs[2][128 * 40];
  const int tileM = blockIdx.y * 128;
  const int tileN = blockIdx.x * 128;
  const int tid = threadIdx.x;
  const int wave = tid >> 5, lane = tid & 31;
  const int waveM = wave & 3, waveN = wave >> 2;
  const int lh = lane & 15;
  const bool hi = lane >= 16;
  const int aOff = hi ? 8 : 0;    // A-fragment K-chunk base (16-bit A layout)
  const int bOff = hi ? 8 : 0;    // B-fragment K-chunk base (same as A)
  const int srow = tid >> 1, shalf = tid & 1;

  // Per-thread staging: 32B of the A tile + 32B of the B tile per K-step.
  const unsigned aofs0 = (unsigned)(((tileM + srow) * K + shalf * 16) * 2);
  const unsigned bofs0 = (unsigned)(((tileN + srow) * K + shalf * 16) * 2);
  unsigned ldsA[2], ldsB[2];
#pragma unroll
  for (int p = 0; p < 2; ++p) {
    ldsA[p] = lds_off(&As[p][srow * 40 + shalf * 16]);
    ldsB[p] = lds_off(&Bs[p][srow * 40 + shalf * 16]);
  }

  f32x8 acc[2][4];
#pragma unroll
  for (int sm = 0; sm < 2; ++sm)
#pragma unroll
    for (int sn = 0; sn < 4; ++sn) acc[sm][sn] = fzero8();

  // prologue: fill buffer 0
  async_ld_b128(ldsA[0],      aofs0,      A);
  async_ld_b128(ldsA[0] + 16, aofs0 + 16, A);
  async_ld_b128(ldsB[0],      bofs0,      Wt);
  async_ld_b128(ldsB[0] + 16, bofs0 + 16, Wt);
  wait_async0();
  __syncthreads();

  int p = 0;
  for (int k0 = 0; k0 < K; k0 += 32) {
    if (k0 + 32 < K) {                    // prefetch next K-step into buffer p^1
      unsigned go = (unsigned)((k0 + 32) * 2);
      async_ld_b128(ldsA[p ^ 1],      aofs0 + go,      A);
      async_ld_b128(ldsA[p ^ 1] + 16, aofs0 + go + 16, A);
      async_ld_b128(ldsB[p ^ 1],      bofs0 + go,      Wt);
      async_ld_b128(ldsB[p ^ 1] + 16, bofs0 + go + 16, Wt);
    }

    bf16x16 af[2], bf[4];
#pragma unroll
    for (int sm = 0; sm < 2; ++sm) {
      int rA = waveM * 32 + sm * 16 + lh;
      bf16x8 lo = *(const bf16x8*)&As[p][rA * 40 + aOff];
      bf16x8 hi8 = *(const bf16x8*)&As[p][rA * 40 + aOff + 16];
      af[sm] = BCAT(lo, hi8);
    }
#pragma unroll
    for (int sn = 0; sn < 4; ++sn) {
      int rB = waveN * 64 + sn * 16 + lh;
      bf16x8 lo = *(const bf16x8*)&Bs[p][rB * 40 + bOff];
      bf16x8 hi8 = *(const bf16x8*)&Bs[p][rB * 40 + bOff + 16];
      bf[sn] = BCAT(lo, hi8);
    }
#pragma unroll
    for (int sm = 0; sm < 2; ++sm)
#pragma unroll
      for (int sn = 0; sn < 4; ++sn)
        acc[sm][sn] = wmma_bf16f32(af[sm], bf[sn], acc[sm][sn]);

    wait_async0();       // our prefetch landed
    __syncthreads();     // everyone's prefetch landed / reads of buf p done
    p ^= 1;
  }

  // Epilogue. C layout: elem r -> row mBase+r, col = lane%16 within 16-tile.
#pragma unroll
  for (int sm = 0; sm < 2; ++sm) {
    int mBase = tileM + waveM * 32 + sm * 16 + (hi ? 8 : 0);
    size_t orow[8];
    if (EPI == 1) {
#pragma unroll
      for (int r = 0; r < 8; ++r) {
        int m0 = mBase + r;
        int w = m0 / 49, t = m0 % 49;
        int bi = w >> 6, wi = w & 63;
        int wh = wi >> 3, ww = wi & 7;
        int sr = wh * 7 + t / 7, sc = ww * 7 + t % 7;
        int gr = sr + 3; if (gr >= 56) gr -= 56;   // roll(+3): out lands here
        int gc = sc + 3; if (gc >= 56) gc -= 56;
        orow[r] = (size_t)bi * 3136 + gr * 56 + gc;
      }
    } else {
#pragma unroll
      for (int r = 0; r < 8; ++r) orow[r] = (size_t)(mBase + r);
    }
#pragma unroll
    for (int sn = 0; sn < 4; ++sn) {
      int n0 = tileN + waveN * 64 + sn * 16 + lh;
      float bv = bias[n0];
      f32x8 c = acc[sm][sn];
#pragma unroll
      for (int r = 0; r < 8; ++r) {
        float v = c[r] + bv;
        if (EPI == 0) {
          if (n0 < 512) v *= 0.17677669529663687f;   // fold softmax scale into q
          outB[orow[r] * (size_t)N + n0] = (__bf16)v;
        } else if (EPI == 1) {
          size_t idx = orow[r] * 512 + n0;
          outF[idx] = v + resid[idx];
        } else if (EPI == 2) {
          float gelu = 0.5f * v * (1.0f + erff(v * 0.70710678118654752f));
          outB[orow[r] * (size_t)N + n0] = (__bf16)gelu;
        } else {
          size_t idx = orow[r] * 512 + n0;
          outF[idx] += v;
        }
      }
    }
  }
}

// ---------------------------------------------------------------------------
// Windowed attention: one block (128 thr, 4 waves) per (window, head).
// qkv rows: (window,token) x 1536 bf16 = [q(pre-scaled) | k | v].
// Wave m owns score rows [m*16, m*16+16): QK^T (4 WMMA), + comb table,
// softmax (ds_swizzle XOR butterflies), P->LDS bf16, P*V (4 WMMA).
// ---------------------------------------------------------------------------
__global__ __launch_bounds__(128) void attn_kernel(const __bf16* __restrict__ qkv,
                                                   const float* __restrict__ comb,
                                                   __bf16* __restrict__ out) {
  __shared__ alignas(16) __bf16 kbuf[64 * 40];        // tokens(pad 64) x HD32, +8 pad
  __shared__ alignas(16) __bf16 vtbuf[32 * 72];       // HD32 x tokens(pad 64), +8 pad
  __shared__ alignas(16) __bf16 pbuf[4 * 16 * 72];    // per-wave 16 x 64 P tile

  const int blk = blockIdx.x;
  const int w = blk >> 4;          // window 0..2047
  const int head = blk & 15;
  const int tid = threadIdx.x;
  const float* cw = comb + (size_t)((w & 63) * 16 + head) * 4096;

  // ---- stage K (row-major, zero-padded) and V^T into LDS ----
  {
    int token = tid >> 1;
    int half = tid & 1;
    bf16x8 k0v = bzero8(), k1v = bzero8(), v0 = bzero8(), v1 = bzero8();
    if (token < 49) {
      size_t base = ((size_t)w * 49 + token) * 1536 + head * 32 + half * 16;
      const bf16x8* gk = (const bf16x8*)(qkv + base + 512);
      k0v = gk[0]; k1v = gk[1];
      const bf16x8* gv = (const bf16x8*)(qkv + base + 1024);
      v0 = gv[0]; v1 = gv[1];
    }
    *(bf16x8*)&kbuf[token * 40 + half * 16] = k0v;
    *(bf16x8*)&kbuf[token * 40 + half * 16 + 8] = k1v;
#pragma unroll
    for (int i = 0; i < 8; ++i) {
      vtbuf[(half * 16 + i) * 72 + token] = v0[i];
      vtbuf[(half * 16 + 8 + i) * 72 + token] = v1[i];
    }
  }
  __syncthreads();

  const int wave = tid >> 5, lane = tid & 31;
  const int lh = lane & 15;
  const int hiOff = (lane >= 16) ? 8 : 0;   // A-frag K chunk / C-layout row offset
  const int koff = (lane >= 16) ? 8 : 0;    // B-frag K chunk (same as A)

  // ---- Q fragment (rows wave*16+lh, zero for pad rows; q pre-scaled) ----
  bf16x8 qa0 = bzero8(), qa1 = bzero8();
  {
    int qrow = wave * 16 + lh;
    if (qrow < 49) {
      size_t base = ((size_t)w * 49 + qrow) * 1536 + head * 32;
      qa0 = *(const bf16x8*)(qkv + base + hiOff);
      qa1 = *(const bf16x8*)(qkv + base + hiOff + 16);
    }
  }
  bf16x16 qfrag = BCAT(qa0, qa1);

  // ---- S = Q K^T : 4 x (16x16) WMMA tiles, K=32 ----
  f32x8 sAcc[4];
#pragma unroll
  for (int j = 0; j < 4; ++j) {
    bf16x8 b0 = *(const bf16x8*)&kbuf[(j * 16 + lh) * 40 + koff];
    bf16x8 b1 = *(const bf16x8*)&kbuf[(j * 16 + lh) * 40 + koff + 16];
    sAcc[j] = wmma_bf16f32(qfrag, BCAT(b0, b1), fzero8());
  }

  // ---- + (rel-pos bias + shift mask) from precomputed table ----
#pragma unroll
  for (int j = 0; j < 4; ++j) {
    const float* cj = cw + j * 16 + lh + (wave * 16 + hiOff) * 64;
#pragma unroll
    for (int r = 0; r < 8; ++r) sAcc[j][r] += cj[r * 64];
  }

  // ---- softmax over 64 cols: 4 reg tiles x 16-lane XOR butterflies ----
#pragma unroll
  for (int r = 0; r < 8; ++r) {
    float mx = fmaxf(fmaxf(sAcc[0][r], sAcc[1][r]), fmaxf(sAcc[2][r], sAcc[3][r]));
    mx = fmaxf(mx, xorSwz<8>(mx)); mx = fmaxf(mx, xorSwz<4>(mx));
    mx = fmaxf(mx, xorSwz<2>(mx)); mx = fmaxf(mx, xorSwz<1>(mx));
    float e[4], sum = 0.f;
#pragma unroll
    for (int j = 0; j < 4; ++j) { e[j] = __expf(sAcc[j][r] - mx); sum += e[j]; }
    sum += xorSwz<8>(sum); sum += xorSwz<4>(sum);
    sum += xorSwz<2>(sum); sum += xorSwz<1>(sum);
    float inv = 1.0f / sum;
#pragma unroll
    for (int j = 0; j < 4; ++j) sAcc[j][r] = e[j] * inv;
  }

  // ---- P tile -> LDS (C layout -> row-major bf16) ----
  __bf16* prow = &pbuf[wave * 16 * 72];
#pragma unroll
  for (int j = 0; j < 4; ++j)
#pragma unroll
    for (int r = 0; r < 8; ++r)
      prow[(r + hiOff) * 72 + j * 16 + lh] = (__bf16)sAcc[j][r];

  // ---- O = P V : K=64 (2 steps) x 2 HD tiles ----
  f32x8 oAcc[2] = {fzero8(), fzero8()};
#pragma unroll
  for (int ks = 0; ks < 2; ++ks) {
    bf16x8 pa0 = *(const bf16x8*)&prow[lh * 72 + ks * 32 + hiOff];
    bf16x8 pa1 = *(const bf16x8*)&prow[lh * 72 + ks * 32 + hiOff + 16];
    bf16x16 pfrag = BCAT(pa0, pa1);
#pragma unroll
    for (int jn = 0; jn < 2; ++jn) {
      bf16x8 vb0 = *(const bf16x8*)&vtbuf[(jn * 16 + lh) * 72 + ks * 32 + koff];
      bf16x8 vb1 = *(const bf16x8*)&vtbuf[(jn * 16 + lh) * 72 + ks * 32 + koff + 16];
      oAcc[jn] = wmma_bf16f32(pfrag, BCAT(vb0, vb1), oAcc[jn]);
    }
  }

  // ---- store (window,token) x 512 bf16, head slice ----
#pragma unroll
  for (int jn = 0; jn < 2; ++jn)
#pragma unroll
    for (int r = 0; r < 8; ++r) {
      int tok = wave * 16 + r + hiOff;
      if (tok < 49)
        out[((size_t)w * 49 + tok) * 512 + head * 32 + jn * 16 + lh] = (__bf16)oAcc[jn][r];
    }
}

// ---------------------------------------------------------------------------
// Host launcher
// ---------------------------------------------------------------------------
extern "C" void kernel_launch(void* const* d_in, const int* in_sizes, int n_in,
                              void* d_out, int out_size, void* d_ws, size_t ws_size,
                              hipStream_t stream) {
  const float* x      = (const float*)d_in[0];
  const float* n1g    = (const float*)d_in[1];
  const float* n1b    = (const float*)d_in[2];
  const float* qkv_w  = (const float*)d_in[3];
  const float* qkv_b  = (const float*)d_in[4];
  const float* relTab = (const float*)d_in[5];
  const float* proj_w = (const float*)d_in[6];
  const float* proj_b = (const float*)d_in[7];
  const float* n2g    = (const float*)d_in[8];
  const float* n2b    = (const float*)d_in[9];
  const float* fc1_w  = (const float*)d_in[10];
  const float* fc1_b  = (const float*)d_in[11];
  const float* fc2_w  = (const float*)d_in[12];
  const float* fc2_b  = (const float*)d_in[13];
  float* out = (float*)d_out;
  (void)in_sizes; (void)n_in; (void)out_size; (void)ws_size;

  char* ws = (char*)d_ws;
  size_t off = 0;
  auto take = [&](size_t bytes) -> char* {
    off = (off + 255) & ~(size_t)255;
    char* p = ws + off;
    off += bytes;
    return p;
  };
  __bf16* wqkvT  = (__bf16*)take((size_t)1536 * 512 * 2);
  __bf16* wprojT = (__bf16*)take((size_t)512 * 512 * 2);
  __bf16* wfc1T  = (__bf16*)take((size_t)2048 * 512 * 2);
  __bf16* wfc2T  = (__bf16*)take((size_t)512 * 2048 * 2);
  float*  comb   = (float*)take((size_t)1024 * 4096 * 4);      // bias+mask table
  __bf16* bufA   = (__bf16*)take((size_t)100352 * 512 * 2);    // hwin / attn_out / ln2
  __bf16* bufB   = (__bf16*)take((size_t)100352 * 2048 * 2);   // qkv / fc1-gelu

  dim3 tb(16, 16);
  castT_kernel<<<dim3(1536 / 16, 512 / 16), tb, 0, stream>>>(qkv_w, wqkvT, 512, 1536);
  castT_kernel<<<dim3(512 / 16, 512 / 16), tb, 0, stream>>>(proj_w, wprojT, 512, 512);
  castT_kernel<<<dim3(2048 / 16, 512 / 16), tb, 0, stream>>>(fc1_w, wfc1T, 512, 2048);
  castT_kernel<<<dim3(512 / 16, 2048 / 16), tb, 0, stream>>>(fc2_w, wfc2T, 2048, 512);
  biasmask_kernel<<<1024, 256, 0, stream>>>(relTab, comb);

  // LN1 + shift + window partition -> bufA (bf16, window-order rows)
  ln_kernel<0><<<12544, 256, 0, stream>>>(x, n1g, n1b, bufA);
  // QKV GEMM (q pre-scaled) -> bufB
  gemm_bf16<0><<<dim3(12, 784), 256, 0, stream>>>(bufA, wqkvT, qkv_b,
                                                  100352, 1536, 512, bufB, nullptr, nullptr);
  // Attention -> bufA (overwrites hwin, no longer needed)
  attn_kernel<<<32768, 128, 0, stream>>>(bufB, comb, bufA);
  // Proj GEMM + window reverse + roll + residual -> d_out (f32, holds x1)
  gemm_bf16<1><<<dim3(4, 784), 256, 0, stream>>>(bufA, wprojT, proj_b,
                                                 100352, 512, 512, nullptr, out, x);
  // LN2 (natural order) -> bufA
  ln_kernel<1><<<12544, 256, 0, stream>>>(out, n2g, n2b, bufA);
  // FC1 + exact GELU -> bufB
  gemm_bf16<2><<<dim3(16, 784), 256, 0, stream>>>(bufA, wfc1T, fc1_b,
                                                  100352, 2048, 512, bufB, nullptr, nullptr);
  // FC2 + bias, accumulate into d_out (x1 + mlp)
  gemm_bf16<3><<<dim3(4, 784), 256, 0, stream>>>(bufB, wfc2T, fc2_b,
                                                 100352, 512, 2048, nullptr, out, nullptr);
}
